// MAGNODecoder_87651692577274
// MI455X (gfx1250) — matrix operand order repaired and checked by reading, not compiled, not run on hardware
//
#include <hip/hip_runtime.h>
#include <hip/hip_bf16.h>
#include <math.h>

typedef _Float16 half_t;
typedef __attribute__((ext_vector_type(16))) _Float16 v16h;
typedef __attribute__((ext_vector_type(8)))  _Float16 v8h;
typedef __attribute__((ext_vector_type(8)))  float    v8f;

#define NB   2
#define NQ_  8192
#define NY_  4096
#define NE   131072
#define NS   2
#define CIN_ 128
#define H1   256      // kernel-MLP hidden width
#define TE   64       // edges per block tile

__device__ __forceinline__ float gelu_tanh(float x) {
    const float c0 = 0.7978845608028654f;   // sqrt(2/pi)
    const float c1 = 0.044715f;
    float x3 = x * x * x;
    return 0.5f * x * (1.0f + tanhf(c0 * (x + c1 * x3)));
}

// ---------------------------------------------------------------------------
// One-time weight prep: f32 -> f16, transposed to [N][K] so each WMMA B-lane
// (lane = output column) reads 16 contiguous halves (32B) from L2.
// ---------------------------------------------------------------------------
__global__ void prep_weights(const float* __restrict__ Wk2,   // [256][256]
                             const float* __restrict__ Wk3,   // [256][128]
                             half_t* __restrict__ Wk2T,       // [256][256]
                             half_t* __restrict__ Wk3T)       // [128][256]
{
    int i = blockIdx.x * blockDim.x + threadIdx.x;
    if (i < 256 * 256) {
        int n = i >> 8, k = i & 255;
        Wk2T[n * 256 + k] = (half_t)Wk2[k * 256 + n];
    }
    if (i < 128 * 256) {
        int n = i >> 8, k = i & 255;
        Wk3T[n * 256 + k] = (half_t)Wk3[k * 128 + n];
    }
}

// ---------------------------------------------------------------------------
// Per-query softmax scale weights: w = softmax(relu(qc@Ws1+bs1)@Ws2+bs2)
// ---------------------------------------------------------------------------
__global__ void scale_weights(const float* __restrict__ qc,   // [B,NQ,2]
                              const float* __restrict__ Ws1,  // [2][16]
                              const float* __restrict__ bs1,  // [16]
                              const float* __restrict__ Ws2,  // [16][2]
                              const float* __restrict__ bs2,  // [2]
                              float* __restrict__ wbuf)       // [B,NQ,2]
{
    int i = blockIdx.x * blockDim.x + threadIdx.x;
    if (i >= NB * NQ_) return;
    float qx = qc[i * 2 + 0], qy = qc[i * 2 + 1];
    float l0 = bs2[0], l1 = bs2[1];
#pragma unroll
    for (int j = 0; j < 16; ++j) {
        float h = fmaxf(qx * Ws1[j] + qy * Ws1[16 + j] + bs1[j], 0.f);
        l0 += h * Ws2[j * 2 + 0];
        l1 += h * Ws2[j * 2 + 1];
    }
    float m  = fmaxf(l0, l1);
    float e0 = __expf(l0 - m), e1 = __expf(l1 - m);
    float r  = 1.f / (e0 + e1);
    wbuf[i * 2 + 0] = e0 * r;
    wbuf[i * 2 + 1] = e1 * r;
}

// ---------------------------------------------------------------------------
// Main fused edge-MLP kernel (WMMA f16 -> f32 accumulate).
// Block = 256 threads = 8 waves, processes TE=64 edges of one (batch, scale).
// ---------------------------------------------------------------------------
__global__ __launch_bounds__(256) void edge_mlp(
    const float* __restrict__ ltc,     // [NY,2]
    const float* __restrict__ rnd,     // [B,NY,128]
    const float* __restrict__ qc,      // [B,NQ,2]
    const int*   __restrict__ q_idx,   // [S,E]
    const int*   __restrict__ y_idx,   // [S,E]
    const float* __restrict__ Wk1,     // [4][256]
    const float* __restrict__ bk1,     // [256]
    const half_t* __restrict__ Wk2T,   // [256][256] f16 (N-major)
    const float* __restrict__ bk2,     // [256]
    const half_t* __restrict__ Wk3T,   // [128][256] f16 (N-major)
    const float* __restrict__ bk3,     // [128]
    const float* __restrict__ wbuf,    // [B,NQ,2]
    float* __restrict__ dec)           // [B,NQ,128]  (atomic accumulate)
{
    __shared__ __align__(16) half_t sh_h1[TE][H1];   // 32 KB
    __shared__ __align__(16) half_t sh_h2[TE][32];   //  4 KB (one K-slab)
    __shared__ float sh_feat[TE][4];
    __shared__ int   sh_qi[TE];
    __shared__ int   sh_yi[TE];
    __shared__ float sh_w[TE];

    const int tid  = threadIdx.x;
    const int lane = tid & 31;
    const int wave = tid >> 5;

    const int s  = blockIdx.y;
    const int b  = blockIdx.z;
    const int e0 = blockIdx.x * TE;

    // ---- edge metadata + gathered coordinate features ----
    if (tid < TE) {
        int e  = e0 + tid;
        int qi = q_idx[s * NE + e];
        int yi = y_idx[s * NE + e];
        sh_qi[tid] = qi;
        sh_yi[tid] = yi;
        sh_w[tid]  = wbuf[(b * NQ_ + qi) * NS + s];
        sh_feat[tid][0] = qc[((size_t)b * NQ_ + qi) * 2 + 0];
        sh_feat[tid][1] = qc[((size_t)b * NQ_ + qi) * 2 + 1];
        sh_feat[tid][2] = ltc[yi * 2 + 0];
        sh_feat[tid][3] = ltc[yi * 2 + 1];
    }
    __syncthreads();

    // ---- layer 1: K=4, done in VALU; h1 stored f16 row-major ----
    {
        int e = tid >> 2;
        int cbase = (tid & 3) * 64;
        float f0 = sh_feat[e][0], f1 = sh_feat[e][1];
        float f2 = sh_feat[e][2], f3 = sh_feat[e][3];
        for (int c = cbase; c < cbase + 64; ++c) {
            float h = f0 * Wk1[c] + f1 * Wk1[H1 + c] +
                      f2 * Wk1[2 * H1 + c] + f3 * Wk1[3 * H1 + c] + bk1[c];
            sh_h1[e][c] = (half_t)gelu_tanh(h);
        }
    }
    __syncthreads();

    // ---- layers 2+3 fused: 8 rounds of 32 h2-columns each ----
    const int mt   = wave & 3;   // 16-edge row tile owned by this wave
    const int ncol = wave >> 2;  // which 16-col half of the 32-col round slab

    v8f acc3[4] = {};            // persistent layer-3 accumulators (4 N-tiles)

    for (int r = 0; r < 8; ++r) {
        // -- layer 2: this wave makes h2 tile (mt, cols r*32+ncol*16 .. +16) --
        v8f acc2 = {};
#pragma unroll
        for (int ks = 0; ks < 8; ++ks) {
            v16h a;
            {   // A fragment (16x32 f16) from LDS h1
                int row = mt * 16 + (lane & 15);
                int klo = (lane < 16) ? 0 : 16;
                int kb  = ks * 32 + klo;
                v8h c0 = *(const v8h*)&sh_h1[row][kb];
                v8h c1 = *(const v8h*)&sh_h1[row][kb + 8];
#pragma unroll
                for (int i = 0; i < 8; ++i) { a[i] = c0[i]; a[i + 8] = c1[i]; }
            }
            v16h bm;
            {   // B fragment (32x16 f16): lane = column, 16 contiguous K halves
                int col = r * 32 + ncol * 16 + (lane & 15);
                int klo = (lane < 16) ? 0 : 16;
                const v8h* p = (const v8h*)&Wk2T[col * H1 + ks * 32 + klo];
                v8h c0 = p[0], c1 = p[1];
#pragma unroll
                for (int i = 0; i < 8; ++i) { bm[i] = c0[i]; bm[i + 8] = c1[i]; }
            }
            acc2 = __builtin_amdgcn_wmma_f32_16x16x32_f16(
                false, a, false, bm, (short)0, acc2, false, false);
        }
        {   // bias + gelu -> f16 slab in LDS
            int lcol = ncol * 16 + (lane & 15);
            float bias = bk2[r * 32 + lcol];
#pragma unroll
            for (int i = 0; i < 8; ++i) {
                int erow = mt * 16 + i + ((lane < 16) ? 0 : 8);
                sh_h2[erow][lcol] = (half_t)gelu_tanh(acc2[i] + bias);
            }
        }
        __syncthreads();

        // -- layer 3 partial: acc3 += h2_slab @ Wk3T[:, r*32 : r*32+32] --
        {
            v16h a;
            {
                int row = mt * 16 + (lane & 15);
                int klo = (lane < 16) ? 0 : 16;
                v8h c0 = *(const v8h*)&sh_h2[row][klo];
                v8h c1 = *(const v8h*)&sh_h2[row][klo + 8];
#pragma unroll
                for (int i = 0; i < 8; ++i) { a[i] = c0[i]; a[i + 8] = c1[i]; }
            }
#pragma unroll
            for (int i = 0; i < 4; ++i) {
                int nt3 = (wave >> 2) * 4 + i;
                v16h bm;
                int col = nt3 * 16 + (lane & 15);
                int klo = (lane < 16) ? 0 : 16;
                const v8h* p = (const v8h*)&Wk3T[col * H1 + r * 32 + klo];
                v8h c0 = p[0], c1 = p[1];
#pragma unroll
                for (int j = 0; j < 8; ++j) { bm[j] = c0[j]; bm[j + 8] = c1[j]; }
                acc3[i] = __builtin_amdgcn_wmma_f32_16x16x32_f16(
                    false, a, false, bm, (short)0, acc3[i], false, false);
            }
        }
        __syncthreads();
    }

    // ---- epilogue: (acc + bk3) * f_y[yi] * w[qi], atomic scatter to dec ----
#pragma unroll
    for (int i = 0; i < 4; ++i) {
        int nt3 = (wave >> 2) * 4 + i;
        int col = nt3 * 16 + (lane & 15);
        float bias = bk3[col];
#pragma unroll
        for (int j = 0; j < 8; ++j) {
            int erow = mt * 16 + j + ((lane < 16) ? 0 : 8);
            int qi = sh_qi[erow];
            int yi = sh_yi[erow];
            float v = (acc3[i][j] + bias) *
                      rnd[((size_t)b * NY_ + yi) * CIN_ + col] * sh_w[erow];
            __hip_atomic_fetch_add(&dec[((size_t)b * NQ_ + qi) * CIN_ + col], v,
                                   __ATOMIC_RELAXED, __HIP_MEMORY_SCOPE_AGENT);
        }
    }
}

// ---------------------------------------------------------------------------
// Projection MLP: out = gelu(dec @ Wp1 + bp1) @ Wp2 + bp2  (one block per row)
// ---------------------------------------------------------------------------
__global__ __launch_bounds__(128) void project(
    const float* __restrict__ dec,   // [B*NQ,128]
    const float* __restrict__ Wp1,   // [128][256]
    const float* __restrict__ bp1,   // [256]
    const float* __restrict__ Wp2,   // [256][3]
    const float* __restrict__ bp2,   // [3]
    float* __restrict__ out)         // [B*NQ,3]
{
    __shared__ float srow[CIN_];
    __shared__ float spart[3][128];
    int row = blockIdx.x;
    int t   = threadIdx.x;
    srow[t] = dec[(size_t)row * CIN_ + t];
    __syncthreads();
    float p0 = 0.f, p1 = 0.f, p2 = 0.f;
    for (int cc = 0; cc < 2; ++cc) {
        int c = t + cc * 128;
        float h = bp1[c];
        for (int k = 0; k < CIN_; ++k) h += srow[k] * Wp1[k * 256 + c];
        h = gelu_tanh(h);
        p0 += h * Wp2[c * 3 + 0];
        p1 += h * Wp2[c * 3 + 1];
        p2 += h * Wp2[c * 3 + 2];
    }
    spart[0][t] = p0; spart[1][t] = p1; spart[2][t] = p2;
    __syncthreads();
    for (int off = 64; off > 0; off >>= 1) {
        if (t < off) {
            spart[0][t] += spart[0][t + off];
            spart[1][t] += spart[1][t + off];
            spart[2][t] += spart[2][t + off];
        }
        __syncthreads();
    }
    if (t < 3) out[(size_t)row * 3 + t] = spart[t][0] + bp2[t];
}

// ---------------------------------------------------------------------------
extern "C" void kernel_launch(void* const* d_in, const int* in_sizes, int n_in,
                              void* d_out, int out_size, void* d_ws, size_t ws_size,
                              hipStream_t stream) {
    (void)in_sizes; (void)n_in; (void)out_size; (void)ws_size;
    const float* ltc  = (const float*)d_in[0];
    const float* rnd  = (const float*)d_in[1];
    const float* qc   = (const float*)d_in[2];
    const int*  q_idx = (const int*)d_in[3];
    const int*  y_idx = (const int*)d_in[4];
    const float* Wk1  = (const float*)d_in[5];
    const float* bk1  = (const float*)d_in[6];
    const float* Wk2  = (const float*)d_in[7];
    const float* bk2  = (const float*)d_in[8];
    const float* Wk3  = (const float*)d_in[9];
    const float* bk3  = (const float*)d_in[10];
    const float* Ws1  = (const float*)d_in[11];
    const float* bs1  = (const float*)d_in[12];
    const float* Ws2  = (const float*)d_in[13];
    const float* bs2  = (const float*)d_in[14];
    const float* Wp1  = (const float*)d_in[15];
    const float* bp1  = (const float*)d_in[16];
    const float* Wp2  = (const float*)d_in[17];
    const float* bp2  = (const float*)d_in[18];
    float* out = (float*)d_out;

    // workspace layout
    char*  ws   = (char*)d_ws;
    half_t* Wk2T = (half_t*)(ws);                               // 128 KB
    half_t* Wk3T = (half_t*)(ws + 131072);                      //  64 KB
    float*  wbuf = (float*)(ws + 131072 + 65536);               // 128 KB
    float*  dec  = (float*)(ws + 131072 + 65536 + 131072);      //   8 MB

    hipMemsetAsync(dec, 0, (size_t)NB * NQ_ * CIN_ * sizeof(float), stream);
    prep_weights<<<256, 256, 0, stream>>>(Wk2, Wk3, Wk2T, Wk3T);
    scale_weights<<<(NB * NQ_ + 255) / 256, 256, 0, stream>>>(qc, Ws1, bs1, Ws2, bs2, wbuf);

    dim3 grid(NE / TE, NS, NB);   // 2048 x 2 x 2 blocks
    edge_mlp<<<grid, 256, 0, stream>>>(ltc, rnd, qc, q_idx, y_idx,
                                       Wk1, bk1, Wk2T, bk2, Wk3T, bk3, wbuf, dec);

    project<<<NB * NQ_, 128, 0, stream>>>(dec, Wp1, bp1, Wp2, bp2, out);
}
